// AttentionBlock_57062935495501
// MI455X (gfx1250) — matrix operand order repaired and checked by reading, not compiled, not run on hardware
//
#include <hip/hip_runtime.h>

#define Bn  32
#define Cn  256
#define HWn 1024

typedef __attribute__((ext_vector_type(16))) _Float16 v16h;
typedef __attribute__((ext_vector_type(8)))  _Float16 v8h;
typedef __attribute__((ext_vector_type(8)))  float    v8f;

__device__ __forceinline__ v8f wmma_f16(v16h a, v16h b, v8f c) {
  // D = A(16x32 f16) * B(32x16 f16) + C(16x16 f32)
  return __builtin_amdgcn_wmma_f32_16x16x32_f16(false, a, false, b, (short)0, c,
                                                false, false);
}

// A-operand frag: 16x32 tile; p -> element [row0][k0] of row-major [*, ld] f16.
// Lane (l&15) = matrix row; lanes 0-15 hold K {0..7, 16..23}, 16-31 {8..15, 24..31}.
__device__ __forceinline__ v16h load_afrag(const _Float16* p, int ld) {
  int lane = threadIdx.x & 31;
  const _Float16* r = p + (size_t)(lane & 15) * ld + ((lane >> 4) << 3);
  union { v16h v; v8h h[2]; } u;
  u.h[0] = *(const v8h*)(r);
  u.h[1] = *(const v8h*)(r + 16);
  return u.v;
}

// B-operand frag: 32x16 tile whose column n is row n of a row-major [*, ld] f16
// source. Lane (l&15) = column; lanes 0-15 hold K {0..7, 16..23}, 16-31 {8..15, 24..31}.
__device__ __forceinline__ v16h load_bfrag(const _Float16* p, int ld) {
  int lane = threadIdx.x & 31;
  const _Float16* r = p + (size_t)(lane & 15) * ld + ((lane >> 4) << 3);
  union { v16h v; v8h h[2]; } u;
  u.h[0] = *(const v8h*)(r);
  u.h[1] = *(const v8h*)(r + 16);
  return u.v;
}

// A-frag built from f32 LDS probability rows [16][HWn]; k0 = key offset.
__device__ __forceinline__ v16h load_pfrag(const float* sc, int k0) {
  int lane = threadIdx.x & 31;
  const float* rp = sc + (size_t)(lane & 15) * HWn + k0 + ((lane >> 4) << 3);
  v16h f;
#pragma unroll
  for (int i = 0; i < 8; ++i) {
    f[i]     = (_Float16)rp[i];
    f[i + 8] = (_Float16)rp[i + 16];
  }
  return f;
}

// ---------------- Kernel 1: GroupNorm(1 group) stats per batch ----------------
__global__ __launch_bounds__(256) void gn_stats_kernel(const float* __restrict__ x,
                                                       float* __restrict__ stats) {
  int b = blockIdx.x;
  const float* xb = x + (size_t)b * Cn * HWn;
  const int N = Cn * HWn;
  float s = 0.f, s2 = 0.f;
  for (int i = threadIdx.x; i < N; i += 256) {
    float v = xb[i];
    s += v; s2 += v * v;
  }
  __shared__ float rs[256], rq[256];
  rs[threadIdx.x] = s; rq[threadIdx.x] = s2;
  __syncthreads();
  for (int off = 128; off > 0; off >>= 1) {
    if (threadIdx.x < off) {
      rs[threadIdx.x] += rs[threadIdx.x + off];
      rq[threadIdx.x] += rq[threadIdx.x + off];
    }
    __syncthreads();
  }
  if (threadIdx.x == 0) {
    float mean = rs[0] / (float)N;
    float var  = rq[0] / (float)N - mean * mean;
    stats[2 * b]     = mean;
    stats[2 * b + 1] = rsqrtf(var + 1e-6f);
  }
}

// ---------------- Kernel 2: weights f32 -> f16 ----------------
__global__ __launch_bounds__(256) void cvt_w_kernel(const float* qw, const float* kw,
                                                    const float* vw, const float* ow,
                                                    _Float16* wq, _Float16* wk,
                                                    _Float16* wv, _Float16* wo) {
  int i = blockIdx.x * 256 + threadIdx.x;  // 65536 total
  wq[i] = (_Float16)qw[i];
  wk[i] = (_Float16)kw[i];
  wv[i] = (_Float16)vw[i];
  wo[i] = (_Float16)ow[i];
}

// ------- Kernel 3: GN apply + NCHW -> (B,HW,C) token transpose, f16 out -------
__global__ __launch_bounds__(256) void norm_tokens_kernel(const float* __restrict__ x,
                                                          const float* __restrict__ gnw,
                                                          const float* __restrict__ gnb,
                                                          const float* __restrict__ stats,
                                                          _Float16* __restrict__ hf) {
  __shared__ float tile[32][33];
  int b  = blockIdx.z;
  int t0 = blockIdx.x * 32;
  int c0 = blockIdx.y * 32;
  float mean = stats[2 * b], rsig = stats[2 * b + 1];
  const float* xb = x + (size_t)b * Cn * HWn;
  int tt = threadIdx.x & 31;
  int cr = threadIdx.x >> 5;  // 8 channel rows per pass
#pragma unroll
  for (int i = 0; i < 4; ++i) {
    int c = cr + i * 8;
    tile[c][tt] = xb[(size_t)(c0 + c) * HWn + t0 + tt];
  }
  __syncthreads();
#pragma unroll
  for (int i = 0; i < 4; ++i) {
    int tl = (threadIdx.x >> 5) + i * 8;
    int c  = c0 + (threadIdx.x & 31);
    float v = (tile[threadIdx.x & 31][tl] - mean) * rsig * gnw[c] + gnb[c];
    hf[((size_t)b * HWn + t0 + tl) * Cn + c] = (_Float16)v;
  }
}

// ---------------- Kernel 4: QKV projections (WMMA) ----------------
__device__ __forceinline__ void proj_store(const v16h* af, const _Float16* w,
                                           const float* bias, float scale,
                                           _Float16* dst, int m0, int trans) {
  int lane = threadIdx.x & 31;
  for (int nt = 0; nt < 16; ++nt) {
    v8f acc = {};
    const _Float16* wr = w + nt * 16 * Cn;
#pragma unroll
    for (int kk = 0; kk < 8; ++kk)
      acc = wmma_f16(af[kk], load_bfrag(wr + kk * 32, Cn), acc);
    int n  = nt * 16 + (lane & 15);
    int mb = m0 + ((lane >> 4) << 3);
    float bv = bias[n];
#pragma unroll
    for (int r = 0; r < 8; ++r) {
      float ov = (acc[r] + bv) * scale;
      if (trans) dst[(size_t)n * HWn + mb + r] = (_Float16)ov;        // [C][HW]
      else       dst[(size_t)(mb + r) * Cn + n] = (_Float16)ov;       // [HW][C]
    }
  }
}

__global__ __launch_bounds__(256) void qkv_kernel(const _Float16* __restrict__ hf,
                                                  const _Float16* __restrict__ wq,
                                                  const _Float16* __restrict__ wk,
                                                  const _Float16* __restrict__ wv,
                                                  const float* __restrict__ qb,
                                                  const float* __restrict__ kb,
                                                  const float* __restrict__ vb,
                                                  _Float16* __restrict__ q,
                                                  _Float16* __restrict__ k,
                                                  _Float16* __restrict__ vT) {
  int b    = blockIdx.y;
  int wave = threadIdx.x >> 5;
  int m0   = blockIdx.x * 128 + wave * 16;
  const _Float16* A = hf + ((size_t)b * HWn + m0) * Cn;
  v16h af[8];
#pragma unroll
  for (int kk = 0; kk < 8; ++kk) af[kk] = load_afrag(A + kk * 32, Cn);
  const float scale = 0.0625f;  // C^-0.5, folded into Q
  proj_store(af, wq, qb, scale, q  + (size_t)b * HWn * Cn, m0, 0);
  proj_store(af, wk, kb, 1.0f,  k  + (size_t)b * HWn * Cn, m0, 0);
  proj_store(af, wv, vb, 1.0f,  vT + (size_t)b * Cn * HWn, m0, 1);
}

// ------- Kernel 5: attention: S = QK^T (scaled), softmax, O = P V -------
__global__ __launch_bounds__(256) void attn_kernel(const _Float16* __restrict__ q,
                                                   const _Float16* __restrict__ k,
                                                   const _Float16* __restrict__ vT,
                                                   _Float16* __restrict__ o) {
  __shared__ float sc[16 * HWn];  // 64 KB: 16 query rows x 1024 key cols, f32
  int b    = blockIdx.y;
  int m0   = blockIdx.x * 16;
  int wave = threadIdx.x >> 5;
  int lane = threadIdx.x & 31;
  const _Float16* qp = q  + ((size_t)b * HWn + m0) * Cn;
  const _Float16* kp = k  + (size_t)b * HWn * Cn;
  const _Float16* vp = vT + (size_t)b * Cn * HWn;

  v16h aq[8];
#pragma unroll
  for (int kk = 0; kk < 8; ++kk) aq[kk] = load_afrag(qp + kk * 32, Cn);

  // --- scores: each wave does 8 of the 64 key-column tiles ---
  for (int i = 0; i < 8; ++i) {
    int ct = wave * 8 + i;
    v8f acc = {};
#pragma unroll
    for (int kk = 0; kk < 8; ++kk)
      acc = wmma_f16(aq[kk], load_bfrag(kp + (size_t)(ct * 16) * Cn + kk * 32, Cn), acc);
    int col = ct * 16 + (lane & 15);
    int rb  = (lane >> 4) << 3;
#pragma unroll
    for (int r = 0; r < 8; ++r) sc[(size_t)(rb + r) * HWn + col] = acc[r];
  }
  __syncthreads();

  // --- softmax: wave handles 2 of the 16 rows, normalized in place (f32) ---
  for (int rr = 0; rr < 2; ++rr) {
    float* srow = sc + (size_t)(wave * 2 + rr) * HWn;
    float m = -3.40282347e38f;
    for (int j = lane; j < HWn; j += 32) m = fmaxf(m, srow[j]);
#pragma unroll
    for (int off = 16; off > 0; off >>= 1) m = fmaxf(m, __shfl_xor(m, off, 32));
    float s = 0.f;
    for (int j = lane; j < HWn; j += 32) {
      float e = __expf(srow[j] - m);
      srow[j] = e;
      s += e;
    }
#pragma unroll
    for (int off = 16; off > 0; off >>= 1) s += __shfl_xor(s, off, 32);
    float inv = 1.0f / s;
    for (int j = lane; j < HWn; j += 32) srow[j] *= inv;
  }
  __syncthreads();

  // --- O = P V: each wave does 2 of the 16 channel tiles ---
  for (int rr = 0; rr < 2; ++rr) {
    int nt = wave * 2 + rr;
    v8f acc = {};
    for (int jc = 0; jc < 32; ++jc) {
      v16h ap = load_pfrag(sc, jc * 32);
      v16h bv = load_bfrag(vp + (size_t)(nt * 16) * HWn + jc * 32, HWn);
      acc = wmma_f16(ap, bv, acc);
    }
    int n  = nt * 16 + (lane & 15);
    int mb = (lane >> 4) << 3;
#pragma unroll
    for (int r = 0; r < 8; ++r)
      o[((size_t)b * HWn + m0 + mb + r) * Cn + n] = (_Float16)acc[r];
  }
}

// ------- Kernel 6: out projection + bias + residual, writes NCHW f32 -------
__global__ __launch_bounds__(256) void outproj_kernel(const _Float16* __restrict__ o,
                                                      const _Float16* __restrict__ wo,
                                                      const float* __restrict__ obias,
                                                      const float* __restrict__ x,
                                                      float* __restrict__ out) {
  int b    = blockIdx.y;
  int wave = threadIdx.x >> 5;
  int lane = threadIdx.x & 31;
  int m0   = blockIdx.x * 128 + wave * 16;
  const _Float16* A = o + ((size_t)b * HWn + m0) * Cn;
  v16h af[8];
#pragma unroll
  for (int kk = 0; kk < 8; ++kk) af[kk] = load_afrag(A + kk * 32, Cn);
  for (int nt = 0; nt < 16; ++nt) {
    v8f acc = {};
#pragma unroll
    for (int kk = 0; kk < 8; ++kk)
      acc = wmma_f16(af[kk], load_bfrag(wo + (size_t)(nt * 16) * Cn + kk * 32, Cn), acc);
    int n  = nt * 16 + (lane & 15);
    int mb = m0 + ((lane >> 4) << 3);
    float bv = obias[n];
    size_t base = (size_t)b * Cn * HWn + (size_t)n * HWn;
#pragma unroll
    for (int r = 0; r < 8; ++r)
      out[base + mb + r] = x[base + mb + r] + acc[r] + bv;
  }
}

// ---------------------------------------------------------------------------
extern "C" void kernel_launch(void* const* d_in, const int* in_sizes, int n_in,
                              void* d_out, int out_size, void* d_ws, size_t ws_size,
                              hipStream_t stream) {
  (void)in_sizes; (void)n_in; (void)out_size; (void)ws_size;
  const float* x    = (const float*)d_in[0];
  // d_in[1] = emb, d_in[2] = cond: unused by reference (has_cond=False)
  const float* gn_w = (const float*)d_in[3];
  const float* gn_b = (const float*)d_in[4];
  const float* qw   = (const float*)d_in[5];
  const float* qb   = (const float*)d_in[6];
  const float* kw   = (const float*)d_in[7];
  const float* kb   = (const float*)d_in[8];
  const float* vw   = (const float*)d_in[9];
  const float* vb   = (const float*)d_in[10];
  const float* ow   = (const float*)d_in[11];
  const float* ob   = (const float*)d_in[12];
  float* out = (float*)d_out;

  char* ws = (char*)d_ws;
  const size_t tok_bytes = (size_t)Bn * HWn * Cn * sizeof(_Float16);  // 16 MB
  const size_t w_bytes   = (size_t)Cn * Cn * sizeof(_Float16);        // 128 KB
  float*    stats = (float*)ws;                                  size_t off = 1024;
  _Float16* hf    = (_Float16*)(ws + off);  off += tok_bytes;    // tokens, reused as O
  _Float16* qbuf  = (_Float16*)(ws + off);  off += tok_bytes;
  _Float16* kbuf  = (_Float16*)(ws + off);  off += tok_bytes;
  _Float16* vTbuf = (_Float16*)(ws + off);  off += tok_bytes;
  _Float16* wq    = (_Float16*)(ws + off);  off += w_bytes;
  _Float16* wk    = (_Float16*)(ws + off);  off += w_bytes;
  _Float16* wv    = (_Float16*)(ws + off);  off += w_bytes;
  _Float16* wo    = (_Float16*)(ws + off);  off += w_bytes;      // ~64.5 MB total

  gn_stats_kernel<<<Bn, 256, 0, stream>>>(x, stats);
  cvt_w_kernel<<<(Cn * Cn) / 256, 256, 0, stream>>>(qw, kw, vw, ow, wq, wk, wv, wo);
  norm_tokens_kernel<<<dim3(HWn / 32, Cn / 32, Bn), 256, 0, stream>>>(x, gn_w, gn_b,
                                                                      stats, hf);
  qkv_kernel<<<dim3(HWn / 128, Bn), 256, 0, stream>>>(hf, wq, wk, wv, qb, kb, vb,
                                                      qbuf, kbuf, vTbuf);
  attn_kernel<<<dim3(HWn / 16, Bn), 256, 0, stream>>>(qbuf, kbuf, vTbuf, hf);
  outproj_kernel<<<dim3(HWn / 128, Bn), 256, 0, stream>>>(hf, wo, ob, x, out);
}
